// IBT_partseg_33028298506412
// MI455X (gfx1250) — compile-verified, matrix operand order checked by reading, not run on hardware
//
#include <hip/hip_runtime.h>
#include <hip/hip_bf16.h>

// ---------------------------------------------------------------------------
// MI455X (gfx1250) implementation. All conv1d/conv2d/attention contractions go
// through one generic WMMA GEMM (f16 inputs staged in LDS, f32 accumulate via
// v_wmma_f32_16x16x32_f16, wave32, 8 waves/block, 64x128 block tile, 32x32 per
// wave) with a fused epilogue: out = res + act((acc + b1[m])*g[m] + b2[m]).
// Staging uses 128-bit global loads + packed f16 LDS stores on the (common)
// fully-in-bounds path; fragments are assembled with 2x ds_load_b128 each.
// ---------------------------------------------------------------------------

typedef __attribute__((ext_vector_type(16))) _Float16 v16h;
typedef __attribute__((ext_vector_type(8)))  _Float16 v8h;
typedef __attribute__((ext_vector_type(4)))  _Float16 v4h;
typedef __attribute__((ext_vector_type(8)))  float    v8f;

#define NPTS 2048
#define KNN  40
#define NBAT 4
#define NKF  (NPTS * KNN)   // 81920

#define BM 64
#define BN 128
#define BK 32
#define APAD 40             // A row stride in f16 (80 B, 16B aligned)
#define BPAD 40             // B^T row stride in f16 (80 B, 16B aligned)

#define ACT_NONE  0
#define ACT_LRELU 1
#define ACT_RELU  2

union Frag { v16h v; v8h h[2]; };

// ---------------------------------------------------------------------------
// Generic WMMA GEMM: C[b, m, col] = res + act((sum_k A(m,k)*B(k,col) + b1)*g + b2)
//   A element:  A[b*sA + m*lda_m + k*lda_k]
//   B element:  B[b*sB + k*ldb_k + col]
//   C element:  C[b*sC + m*ldc_m + col]
//   res element:res[b*sR + m*ldr_m + col]   (optional, added after activation)
// ---------------------------------------------------------------------------
__global__ __launch_bounds__(256)
void k_gemm_wmma(const float* __restrict__ A, long lda_m, long lda_k, long sA,
                 const float* __restrict__ Bm, long ldb_k, long sB,
                 float* __restrict__ C, long ldc_m, long sC,
                 const float* __restrict__ b1v,
                 const float* __restrict__ gv,
                 const float* __restrict__ b2v,
                 const float* __restrict__ res, long ldr_m, long sR,
                 int M, int Ncols, int Kd, int act)
{
    __shared__ _Float16 As[BM][APAD];      // [row][k]
    __shared__ _Float16 Bs[BN][BPAD];      // [col][k]  (transposed tile)

    const int tid  = threadIdx.x;
    const int lane = tid & 31;
    const int wave = tid >> 5;      // 0..7
    const int wm   = wave >> 2;     // 0..1  (M direction)
    const int wn   = wave & 3;      // 0..3  (N direction)
    const int half = lane >> 4;     // K-half selector per ISA layout
    const int l16  = lane & 15;

    const int tileN = blockIdx.x * BN;
    const int tileM = blockIdx.y * BM;
    const int b     = blockIdx.z;

    const float* Ab = A  + (long)b * sA;
    const float* Bb = Bm + (long)b * sB;

    // block-uniform fast-path eligibility
    const bool fullM = (tileM + BM <= M);
    const bool fullN = (tileN + BN <= Ncols);
    const bool aVec  = fullM && (lda_k == 1) && ((lda_m & 3) == 0);
    const bool bVec  = fullN && ((ldb_k & 3) == 0);

    v8f acc[2][2];
    #pragma unroll
    for (int i = 0; i < 2; ++i)
        #pragma unroll
        for (int j = 0; j < 2; ++j)
            #pragma unroll
            for (int e = 0; e < 8; ++e) acc[i][j][e] = 0.0f;

    for (int k0 = 0; k0 < Kd; k0 += BK) {
        const bool fullK = (k0 + BK <= Kd);

        // ---- stage A tile (BM x BK) as f16: 2 x float4 per thread ----
        if (aVec && fullK) {
            #pragma unroll
            for (int t = 0; t < 2; ++t) {
                int idx = tid + t * 256;            // 0..511 float4 slots
                int r = idx >> 3, c4 = idx & 7;
                const float4 f = *(const float4*)(Ab + (long)(tileM + r) * lda_m
                                                  + k0 + c4 * 4);
                v4h hv;
                hv[0] = (_Float16)f.x; hv[1] = (_Float16)f.y;
                hv[2] = (_Float16)f.z; hv[3] = (_Float16)f.w;
                *(v4h*)&As[r][c4 * 4] = hv;
            }
        } else {
            #pragma unroll
            for (int t = 0; t < (BM * BK) / 256; ++t) {
                int idx = tid + t * 256;
                int r = idx >> 5, c = idx & 31;
                int m = tileM + r, k = k0 + c;
                float v = 0.0f;
                if (m < M && k < Kd) v = Ab[(long)m * lda_m + (long)k * lda_k];
                As[r][c] = (_Float16)v;
            }
        }
        // ---- stage B tile (BK x BN) as f16, transposed: 4 x float4 per thread
        if (bVec && fullK) {
            #pragma unroll
            for (int t = 0; t < 4; ++t) {
                int idx = tid + t * 256;            // 0..1023 float4 slots
                int k = idx >> 5, c4 = idx & 31;
                const float4 f = *(const float4*)(Bb + (long)(k0 + k) * ldb_k
                                                  + tileN + c4 * 4);
                Bs[c4 * 4 + 0][k] = (_Float16)f.x;
                Bs[c4 * 4 + 1][k] = (_Float16)f.y;
                Bs[c4 * 4 + 2][k] = (_Float16)f.z;
                Bs[c4 * 4 + 3][k] = (_Float16)f.w;
            }
        } else {
            #pragma unroll
            for (int t = 0; t < (BK * BN) / 256; ++t) {
                int idx = tid + t * 256;
                int k = idx >> 7, c = idx & 127;
                int col = tileN + c, kk = k0 + k;
                float v = 0.0f;
                if (col < Ncols && kk < Kd) v = Bb[(long)kk * ldb_k + col];
                Bs[c][k] = (_Float16)v;
            }
        }
        __syncthreads();

        // ---- fragments: e=0..7 -> K=8*half+e ; e=8..15 -> K=16+8*half+(e-8)
        // Both are contiguous 16-byte runs -> 2x ds_load_b128 per fragment.
        Frag af[2], bf[2];
        #pragma unroll
        for (int i = 0; i < 2; ++i) {
            int rowA = wm * 32 + i * 16 + l16;
            af[i].h[0] = *(const v8h*)&As[rowA][8 * half];
            af[i].h[1] = *(const v8h*)&As[rowA][16 + 8 * half];
        }
        #pragma unroll
        for (int j = 0; j < 2; ++j) {
            int colB = wn * 32 + j * 16 + l16;
            bf[j].h[0] = *(const v8h*)&Bs[colB][8 * half];
            bf[j].h[1] = *(const v8h*)&Bs[colB][16 + 8 * half];
        }
        #pragma unroll
        for (int i = 0; i < 2; ++i)
            #pragma unroll
            for (int j = 0; j < 2; ++j)
                acc[i][j] = __builtin_amdgcn_wmma_f32_16x16x32_f16(
                    false, af[i].v, false, bf[j].v, (short)0, acc[i][j],
                    false, false);
        __syncthreads();
    }

    // ---- epilogue: D layout lane l, vgpr v -> m = v + 8*(l>>4), n = l&15 ----
    #pragma unroll
    for (int i = 0; i < 2; ++i) {
        #pragma unroll
        for (int v = 0; v < 8; ++v) {
            int m = tileM + wm * 32 + i * 16 + v + 8 * half;
            bool mok = (m < M);
            float bb1 = (mok && b1v) ? b1v[m] : 0.0f;
            float gg  = (mok && gv)  ? gv[m]  : 1.0f;
            float bb2 = (mok && b2v) ? b2v[m] : 0.0f;
            #pragma unroll
            for (int j = 0; j < 2; ++j) {
                int col = tileN + wn * 32 + j * 16 + l16;
                if (mok && col < Ncols) {
                    float val = acc[i][j][v];
                    val = (val + bb1) * gg + bb2;
                    if (act == ACT_LRELU)      val = val > 0.0f ? val : 0.2f * val;
                    else if (act == ACT_RELU)  val = fmaxf(val, 0.0f);
                    if (res) val += res[(long)b * sR + (long)m * ldr_m + col];
                    C[(long)b * sC + (long)m * ldc_m + col] = val;
                }
            }
        }
    }
}

// ---------------------------------------------------------------------------
// KNN: pd = -||pi-pj||^2 ; keep top-40 (largest pd) per point, sorted desc.
// ---------------------------------------------------------------------------
__global__ void k_knn(const float* __restrict__ xyz, int* __restrict__ idx,
                      float* __restrict__ dis)
{
    int t = blockIdx.x * blockDim.x + threadIdx.x;
    if (t >= NBAT * NPTS) return;
    int b = t / NPTS, n = t % NPTS;
    const float* X = xyz + (long)b * 3 * NPTS;
    float px = X[n], py = X[NPTS + n], pz = X[2 * NPTS + n];
    float sqn = px * px + py * py + pz * pz;
    float best[KNN]; int bidx[KNN];
    for (int i = 0; i < KNN; ++i) { best[i] = -1e30f; bidx[i] = 0; }
    for (int m = 0; m < NPTS; ++m) {
        float qx = X[m], qy = X[NPTS + m], qz = X[2 * NPTS + m];
        float pd = 2.0f * (px * qx + py * qy + pz * qz) - sqn
                 - (qx * qx + qy * qy + qz * qz);
        if (pd > best[KNN - 1]) {
            int p = KNN - 1;
            while (p > 0 && best[p - 1] < pd) {
                best[p] = best[p - 1]; bidx[p] = bidx[p - 1]; --p;
            }
            best[p] = pd; bidx[p] = m;
        }
    }
    for (int i = 0; i < KNN; ++i) {
        idx[(long)t * KNN + i] = bidx[i];
        dis[(long)t * KNN + i] = best[i];
    }
}

// Build pos132 (132, N*K) and xg (128, N*K) for one batch.
__global__ void k_build(const float* __restrict__ xyz_b,
                        const float* __restrict__ feat, long fld,
                        const int* __restrict__ idx_b,
                        const float* __restrict__ dis_b,
                        float* __restrict__ pos132,
                        float* __restrict__ xg)
{
    int t = blockIdx.x * blockDim.x + threadIdx.x;
    if (t >= NKF) return;
    int n = t / KNN;
    int i = idx_b[t];
    const long NK = (long)NKF;
    pos132[0 * NK + t] = xyz_b[n]            - xyz_b[i];
    pos132[1 * NK + t] = xyz_b[NPTS + n]     - xyz_b[NPTS + i];
    pos132[2 * NK + t] = xyz_b[2 * NPTS + n] - xyz_b[2 * NPTS + i];
    pos132[3 * NK + t] = dis_b[t];
    for (int c = 0; c < 128; ++c) {
        float d = feat[(long)c * fld + i] - feat[(long)c * fld + n];
        pos132[(4 + c) * NK + t] = d;
        xg[(long)c * NK + t]     = d;
    }
}

// max over K neighbors: (128, N*K) -> (128, N)
__global__ void k_hmax(const float* __restrict__ h, float* __restrict__ out)
{
    int t = blockIdx.x * blockDim.x + threadIdx.x;
    if (t >= 128 * NPTS) return;
    int c = t / NPTS, n = t % NPTS;
    const float* p = h + (long)c * NKF + (long)n * KNN;
    float m = p[0];
    for (int k = 1; k < KNN; ++k) m = fmaxf(m, p[k]);
    out[(long)c * NPTS + n] = m;
}

// softmax over K of s, weighted sum with h: (128,N*K)x2 -> (128,N)
__global__ void k_scoreatt(const float* __restrict__ s, const float* __restrict__ h,
                           float* __restrict__ out)
{
    int t = blockIdx.x * blockDim.x + threadIdx.x;
    if (t >= 128 * NPTS) return;
    int d = t / NPTS, n = t % NPTS;
    const float* sp = s + (long)d * NKF + (long)n * KNN;
    const float* hp = h + (long)d * NKF + (long)n * KNN;
    float mx = sp[0];
    for (int k = 1; k < KNN; ++k) mx = fmaxf(mx, sp[k]);
    float e[KNN], sum = 0.0f;
    for (int k = 0; k < KNN; ++k) { e[k] = __expf(sp[k] - mx); sum += e[k]; }
    float r = 0.0f;
    for (int k = 0; k < KNN; ++k) r += e[k] * hp[k];
    out[(long)d * NPTS + n] = r / sum;
}

// generic per-batch elementwise: o = a +/- c  (batch strides differ)
__global__ void k_ew2(const float* __restrict__ a, long sa_b,
                      const float* __restrict__ c, long sc_b,
                      float* __restrict__ o, long so_b,
                      long per_b, int nb, int op)
{
    long t = blockIdx.x * (long)blockDim.x + threadIdx.x;
    if (t >= per_b * nb) return;
    int b = (int)(t / per_b); long f = t % per_b;
    float x = a[b * sa_b + f], y = c[b * sc_b + f];
    o[b * so_b + f] = (op == 0) ? (x + y) : (x - y);
}

// v = (vraw + pos) * sigmoid(x_st), all (B,128,N) contiguous
__global__ void k_vfuse(const float* __restrict__ vr, const float* __restrict__ pos,
                        const float* __restrict__ xs, float* __restrict__ o, long tot)
{
    long t = blockIdx.x * (long)blockDim.x + threadIdx.x;
    if (t >= tot) return;
    float s = 1.0f / (1.0f + __expf(-xs[t]));
    o[t] = (vr[t] + pos[t]) * s;
}

// row softmax over last dim of att (B,N,N), in place
__global__ void k_softmax_row(float* __restrict__ att)
{
    __shared__ float sm[256];
    const int tid = threadIdx.x;
    float* p = att + (long)blockIdx.y * NPTS * NPTS + (long)blockIdx.x * NPTS;
    float m = -1e30f;
    for (int i = tid; i < NPTS; i += 256) m = fmaxf(m, p[i]);
    sm[tid] = m; __syncthreads();
    for (int s = 128; s > 0; s >>= 1) {
        if (tid < s) sm[tid] = fmaxf(sm[tid], sm[tid + s]);
        __syncthreads();
    }
    float mx = sm[0]; __syncthreads();
    float sum = 0.0f;
    for (int i = tid; i < NPTS; i += 256) {
        float e = __expf(p[i] - mx); p[i] = e; sum += e;
    }
    sm[tid] = sum; __syncthreads();
    for (int s = 128; s > 0; s >>= 1) {
        if (tid < s) sm[tid] += sm[tid + s];
        __syncthreads();
    }
    float inv = 1.0f / sm[0];
    for (int i = tid; i < NPTS; i += 256) p[i] *= inv;
}

__global__ void k_colsum(const float* __restrict__ att, float* __restrict__ cs)
{
    int t = blockIdx.x * blockDim.x + threadIdx.x;
    if (t >= NBAT * NPTS) return;
    int b = t / NPTS, m = t % NPTS;
    const float* p = att + (long)b * NPTS * NPTS + m;
    float s = 0.0f;
    for (int n = 0; n < NPTS; ++n) s += p[(long)n * NPTS];
    cs[t] = s;
}

__global__ void k_colnorm(float* __restrict__ att, const float* __restrict__ cs)
{
    long t = blockIdx.x * (long)blockDim.x + threadIdx.x;
    long tot = (long)NBAT * NPTS * NPTS;
    if (t >= tot) return;
    int m = (int)(t & (NPTS - 1));
    int b = (int)(t >> 22);               // NPTS*NPTS == 2^22
    att[t] /= (1e-9f + cs[b * NPTS + m]);
}

// max over N: in (B,C,N) -> outb either broadcast (B,C,N) or scalar (B,C)
__global__ void k_maxn(const float* __restrict__ in, long sin_b,
                       float* __restrict__ outb, long sout_b, int bcast)
{
    __shared__ float sm[256];
    int b = blockIdx.y, c = blockIdx.x;
    const float* p = in + (long)b * sin_b + (long)c * NPTS;
    float m = -1e30f;
    for (int n = threadIdx.x; n < NPTS; n += 256) m = fmaxf(m, p[n]);
    sm[threadIdx.x] = m; __syncthreads();
    for (int s = 128; s > 0; s >>= 1) {
        if (threadIdx.x < s) sm[threadIdx.x] = fmaxf(sm[threadIdx.x], sm[threadIdx.x + s]);
        __syncthreads();
    }
    float mx = sm[0];
    if (bcast) {
        for (int n = threadIdx.x; n < NPTS; n += 256)
            outb[(long)b * sout_b + (long)c * NPTS + n] = mx;
    } else if (threadIdx.x == 0) {
        outb[b * sout_b + c] = mx;
    }
}

// lf = lrelu(g*(Wl @ l) + b): (B,64)
__global__ void k_lf(const float* __restrict__ l, const float* __restrict__ W,
                     const float* __restrict__ g, const float* __restrict__ bb,
                     float* __restrict__ lf)
{
    int t = blockIdx.x * blockDim.x + threadIdx.x;
    if (t >= NBAT * 64) return;
    int b = t / 64, o = t % 64;
    float s = 0.0f;
    for (int c = 0; c < 16; ++c) s += W[o * 16 + c] * l[b * 16 + c];
    s = s * g[o] + bb[o];
    lf[t] = s > 0.0f ? s : 0.2f * s;
}

// fill conv8 input (B,1472,N): [hg(1024) ; lf(64)] broadcast over N, then x0,x1,x2
__global__ void k_cat8(const float* __restrict__ hg, const float* __restrict__ lf,
                       const float* __restrict__ cat7, float* __restrict__ cat8)
{
    long t = blockIdx.x * (long)blockDim.x + threadIdx.x;
    long tot = (long)NBAT * 1472 * NPTS;
    if (t >= tot) return;
    int n = (int)(t % NPTS);
    long r = t / NPTS;
    int c = (int)(r % 1472);
    int b = (int)(r / 1472);
    float v;
    if (c < 1024)      v = hg[b * 1024 + c];
    else if (c < 1088) v = lf[b * 64 + (c - 1024)];
    else               v = cat7[(long)b * 512 * NPTS + (long)(c - 1088 + 128) * NPTS + n];
    cat8[t] = v;
}

// ---------------------------------------------------------------------------
// Host-side orchestration
// ---------------------------------------------------------------------------
struct Blk { const float *W, *g, *b; };
struct SAp { const float *Wq, *Wk, *Wv, *bv, *Wt, *bt, *g, *b, *Wpos, *bpos; };

static void gemm(hipStream_t st,
                 const float* A, long lda_m, long lda_k, long sA,
                 const float* Bm, long ldb_k, long sB,
                 float* C, long ldc_m, long sC,
                 const float* b1, const float* g, const float* b2,
                 const float* res, long ldr_m, long sR,
                 int M, int Nc, int Kd, int nb, int act)
{
    dim3 grid((Nc + BN - 1) / BN, (M + BM - 1) / BM, nb);
    k_gemm_wmma<<<grid, dim3(256), 0, st>>>(A, lda_m, lda_k, sA, Bm, ldb_k, sB,
                                            C, ldc_m, sC, b1, g, b2,
                                            res, ldr_m, sR, M, Nc, Kd, act);
}

extern "C" void kernel_launch(void* const* d_in, const int* in_sizes, int n_in,
                              void* d_out, int out_size, void* d_ws, size_t ws_size,
                              hipStream_t stream)
{
    (void)in_sizes; (void)n_in; (void)out_size; (void)ws_size;
    const long Nn = NPTS;
    const long NK = NKF;

    const float* X = (const float*)d_in[0];   // (4,3,2048)
    const float* L = (const float*)d_in[1];   // (4,16,1)

    int p = 2;
    auto nb3 = [&](Blk& k) {
        k.W = (const float*)d_in[p++]; k.g = (const float*)d_in[p++];
        k.b = (const float*)d_in[p++];
    };
    Blk point1, point2, point3, mlpA[3], convA[3], fuseA[3], conv7, convl,
        conv8, conv9, conv10;
    nb3(point1); nb3(point2); nb3(point3);
    nb3(mlpA[0]); nb3(mlpA[1]); nb3(mlpA[2]);
    nb3(convA[0]); nb3(convA[1]); nb3(convA[2]);       // conv0, conv1, conv3
    nb3(fuseA[0]); nb3(fuseA[1]); nb3(fuseA[2]);       // conv4, conv5, conv6
    nb3(conv7); nb3(convl); nb3(conv8); nb3(conv9); nb3(conv10);
    const float* W11 = (const float*)d_in[p++];
    const float* WsA[3];
    for (int i = 0; i < 3; ++i) WsA[i] = (const float*)d_in[p++];
    SAp sa[3];
    for (int i = 0; i < 3; ++i) {
        sa[i].Wq  = (const float*)d_in[p++]; sa[i].Wk   = (const float*)d_in[p++];
        sa[i].Wv  = (const float*)d_in[p++]; sa[i].bv   = (const float*)d_in[p++];
        sa[i].Wt  = (const float*)d_in[p++]; sa[i].bt   = (const float*)d_in[p++];
        sa[i].g   = (const float*)d_in[p++]; sa[i].b    = (const float*)d_in[p++];
        sa[i].Wpos= (const float*)d_in[p++]; sa[i].bpos = (const float*)d_in[p++];
    }

    // ---- workspace carve-out (floats) ----
    float* ws = (float*)d_ws;
    size_t off = 0;
    auto alloc = [&](size_t n) { float* q = ws + off; off += n; return q; };
    float* xp     = alloc((size_t)NBAT * 128 * Nn);
    float* cat7   = alloc((size_t)NBAT * 512 * Nn);   // x00 | x0 | x1 | x2
    int*   knn_i  = (int*)alloc((size_t)NBAT * Nn * KNN);
    float* knn_d  = alloc((size_t)NBAT * Nn * KNN);
    float* pos132 = alloc((size_t)132 * NK);          // per-batch; reused for s_lin
    float* cat192 = alloc((size_t)192 * NK);          // per-batch
    float* h128   = alloc((size_t)128 * NK);          // per-batch
    float* cat256 = alloc((size_t)NBAT * 256 * Nn);
    float* xst    = alloc((size_t)NBAT * 128 * Nn);
    float* possa  = alloc((size_t)NBAT * 128 * Nn);
    float* hsa    = alloc((size_t)NBAT * 128 * Nn);
    float* qbuf   = alloc((size_t)NBAT * 32 * Nn);
    float* kbuf   = alloc((size_t)NBAT * 32 * Nn);
    float* vraw   = alloc((size_t)NBAT * 128 * Nn);
    float* vbuf   = alloc((size_t)NBAT * 128 * Nn);
    float* att    = alloc((size_t)NBAT * Nn * Nn);
    float* colsum = alloc((size_t)NBAT * Nn);
    float* diffb  = alloc((size_t)NBAT * 128 * Nn);
    float* saout  = alloc((size_t)NBAT * 128 * Nn);
    float* h1024  = alloc((size_t)NBAT * 1024 * Nn);
    float* hg     = alloc((size_t)NBAT * 1024);
    float* lf     = alloc((size_t)NBAT * 64);
    float* cat8   = alloc((size_t)NBAT * 1472 * Nn);
    float* t512   = alloc((size_t)NBAT * 512 * Nn);
    float* t256   = alloc((size_t)NBAT * 256 * Nn);
    float* t128   = alloc((size_t)NBAT * 128 * Nn);
    float* h32    = alloc((size_t)NBAT * 32 * Nn);
    float* h64a   = alloc((size_t)NBAT * 64 * Nn);

    // ---- point MLP: 3->32->64->128 (all WMMA GEMMs, bn+lrelu fused) ----
    gemm(stream, point1.W, 3, 1, 0,   X, Nn, 3 * Nn,   h32, Nn, 32 * Nn,
         nullptr, point1.g, point1.b, nullptr, 0, 0, 32, (int)Nn, 3, NBAT, ACT_LRELU);
    gemm(stream, point2.W, 32, 1, 0,  h32, Nn, 32 * Nn, h64a, Nn, 64 * Nn,
         nullptr, point2.g, point2.b, nullptr, 0, 0, 64, (int)Nn, 32, NBAT, ACT_LRELU);
    gemm(stream, point3.W, 64, 1, 0,  h64a, Nn, 64 * Nn, xp, Nn, 128 * Nn,
         nullptr, point3.g, point3.b, nullptr, 0, 0, 128, (int)Nn, 64, NBAT, ACT_LRELU);

    // x00 = broadcast(max_n x_p) into cat7 channels [0,128)
    k_maxn<<<dim3(128, NBAT), 256, 0, stream>>>(xp, 128 * Nn, cat7, 512 * Nn, 1);

    // KNN once (xyz is constant across stages)
    k_knn<<<(NBAT * NPTS + 255) / 256, 256, 0, stream>>>(X, knn_i, knn_d);

    // ---- three stages ----
    for (int st = 0; st < 3; ++st) {
        const float* feat; long fbs;
        if (st == 0) { feat = xp;                      fbs = 128 * Nn; }
        else         { feat = cat7 + (long)(128 * st) * Nn; fbs = 512 * Nn; }
        float* outsl = cat7 + (long)(128 * (st + 1)) * Nn;   // stage output slice

        // per-batch graph feature + conv2d chain
        for (int b = 0; b < NBAT; ++b) {
            k_build<<<(NKF + 255) / 256, 256, 0, stream>>>(
                X + (long)b * 3 * Nn, feat + (long)b * fbs, Nn,
                knn_i + (long)b * Nn * KNN, knn_d + (long)b * Nn * KNN,
                pos132, cat192 + 64 * NK);
            // mlp: 132 -> 64 into cat192[0:64]
            gemm(stream, mlpA[st].W, 132, 1, 0, pos132, NK, 0, cat192, NK, 0,
                 nullptr, mlpA[st].g, mlpA[st].b, nullptr, 0, 0,
                 64, (int)NK, 132, 1, ACT_LRELU);
            // conv: 192 -> 128
            gemm(stream, convA[st].W, 192, 1, 0, cat192, NK, 0, h128, NK, 0,
                 nullptr, convA[st].g, convA[st].b, nullptr, 0, 0,
                 128, (int)NK, 192, 1, ACT_LRELU);
            // hmax -> cat256[b, 0:128]
            k_hmax<<<(128 * NPTS + 255) / 256, 256, 0, stream>>>(
                h128, cat256 + (long)b * 256 * Nn);
            // score linear: Ws @ h  (reuse pos132 as s_lin)
            gemm(stream, WsA[st], 128, 1, 0, h128, NK, 0, pos132, NK, 0,
                 nullptr, nullptr, nullptr, nullptr, 0, 0,
                 128, (int)NK, 128, 1, ACT_NONE);
            // softmax over K + weighted sum -> cat256[b, 128:256]
            k_scoreatt<<<(128 * NPTS + 255) / 256, 256, 0, stream>>>(
                pos132, h128, cat256 + (long)b * 256 * Nn + 128 * Nn);
        }

        // x_st = lrelu(bn(fuse @ cat256))
        gemm(stream, fuseA[st].W, 256, 1, 0, cat256, Nn, 256 * Nn, xst, Nn, 128 * Nn,
             nullptr, fuseA[st].g, fuseA[st].b, nullptr, 0, 0,
             128, (int)Nn, 256, NBAT, ACT_LRELU);

        // ---- SA layer ----
        gemm(stream, sa[st].Wpos, 3, 1, 0, X, Nn, 3 * Nn, possa, Nn, 128 * Nn,
             nullptr, nullptr, sa[st].bpos, nullptr, 0, 0,
             128, (int)Nn, 3, NBAT, ACT_NONE);
        k_ew2<<<(NBAT * 128 * NPTS + 255) / 256, 256, 0, stream>>>(
            feat, fbs, possa, 128 * Nn, hsa, 128 * Nn, 128 * Nn, NBAT, 0);
        gemm(stream, sa[st].Wq, 128, 1, 0, hsa, Nn, 128 * Nn, qbuf, Nn, 32 * Nn,
             nullptr, nullptr, nullptr, nullptr, 0, 0, 32, (int)Nn, 128, NBAT, ACT_NONE);
        gemm(stream, sa[st].Wk, 128, 1, 0, hsa, Nn, 128 * Nn, kbuf, Nn, 32 * Nn,
             nullptr, nullptr, nullptr, nullptr, 0, 0, 32, (int)Nn, 128, NBAT, ACT_NONE);
        gemm(stream, sa[st].Wv, 128, 1, 0, hsa, Nn, 128 * Nn, vraw, Nn, 128 * Nn,
             nullptr, nullptr, sa[st].bv, nullptr, 0, 0, 128, (int)Nn, 128, NBAT, ACT_NONE);
        k_vfuse<<<(NBAT * 128 * NPTS + 255) / 256, 256, 0, stream>>>(
            vraw, possa, xst, vbuf, (long)NBAT * 128 * Nn);
        // energy = Q^T K : A transposed per-batch (lda_m=1, lda_k=N)
        gemm(stream, qbuf, 1, Nn, 32 * Nn, kbuf, Nn, 32 * Nn, att, Nn, Nn * Nn,
             nullptr, nullptr, nullptr, nullptr, 0, 0,
             (int)Nn, (int)Nn, 32, NBAT, ACT_NONE);
        k_softmax_row<<<dim3(NPTS, NBAT), 256, 0, stream>>>(att);
        k_colsum<<<(NBAT * NPTS + 255) / 256, 256, 0, stream>>>(att, colsum);
        {
            long tot = (long)NBAT * NPTS * NPTS;
            k_colnorm<<<(unsigned)((tot + 255) / 256), 256, 0, stream>>>(att, colsum);
        }
        // xr = V @ att : K-dim 2048
        gemm(stream, vbuf, Nn, 1, 128 * Nn, att, Nn, Nn * Nn, vraw, Nn, 128 * Nn,
             nullptr, nullptr, nullptr, nullptr, 0, 0,
             128, (int)Nn, (int)Nn, NBAT, ACT_NONE);
        // diff = feat - xr ; saout = feat + relu(bn(Wt@diff + bt))
        k_ew2<<<(NBAT * 128 * NPTS + 255) / 256, 256, 0, stream>>>(
            feat, fbs, vraw, 128 * Nn, diffb, 128 * Nn, 128 * Nn, NBAT, 1);
        gemm(stream, sa[st].Wt, 128, 1, 0, diffb, Nn, 128 * Nn, saout, Nn, 128 * Nn,
             sa[st].bt, sa[st].g, sa[st].b, feat, Nn, fbs,
             128, (int)Nn, 128, NBAT, ACT_RELU);
        // stage output = x_st + sa_out -> cat7 slice
        k_ew2<<<(NBAT * 128 * NPTS + 255) / 256, 256, 0, stream>>>(
            xst, 128 * Nn, saout, 128 * Nn, outsl, 512 * Nn, 128 * Nn, NBAT, 0);
    }

    // ---- head ----
    gemm(stream, conv7.W, 512, 1, 0, cat7, Nn, 512 * Nn, h1024, Nn, 1024 * Nn,
         nullptr, conv7.g, conv7.b, nullptr, 0, 0, 1024, (int)Nn, 512, NBAT, ACT_LRELU);
    k_maxn<<<dim3(1024, NBAT), 256, 0, stream>>>(h1024, 1024 * Nn, hg, 1024, 0);
    k_lf<<<1, 256, 0, stream>>>(L, convl.W, convl.g, convl.b, lf);
    {
        long tot = (long)NBAT * 1472 * Nn;
        k_cat8<<<(unsigned)((tot + 255) / 256), 256, 0, stream>>>(hg, lf, cat7, cat8);
    }
    gemm(stream, conv8.W, 1472, 1, 0, cat8, Nn, 1472 * Nn, t512, Nn, 512 * Nn,
         nullptr, conv8.g, conv8.b, nullptr, 0, 0, 512, (int)Nn, 1472, NBAT, ACT_LRELU);
    gemm(stream, conv9.W, 512, 1, 0, t512, Nn, 512 * Nn, t256, Nn, 256 * Nn,
         nullptr, conv9.g, conv9.b, nullptr, 0, 0, 256, (int)Nn, 512, NBAT, ACT_LRELU);
    gemm(stream, conv10.W, 256, 1, 0, t256, Nn, 256 * Nn, t128, Nn, 128 * Nn,
         nullptr, conv10.g, conv10.b, nullptr, 0, 0, 128, (int)Nn, 256, NBAT, ACT_LRELU);
    gemm(stream, W11, 128, 1, 0, t128, Nn, 128 * Nn, (float*)d_out, Nn, 50 * Nn,
         nullptr, nullptr, nullptr, nullptr, 0, 0, 50, (int)Nn, 128, NBAT, ACT_NONE);
}
